// ClassicalAttention_65481071397493
// MI455X (gfx1250) — matrix operand order repaired and checked by reading, not compiled, not run on hardware
//
#include <hip/hip_runtime.h>

// Problem sizes (from reference): inputs (S,B,E) f32, params (E,E) f32.
#define SQ 512
#define BQ 512
#define EQ 768

typedef __attribute__((ext_vector_type(16))) __bf16 v16bf;
typedef __attribute__((ext_vector_type(8)))  float  v8f;
typedef unsigned short u16;
typedef __attribute__((ext_vector_type(8)))  unsigned short u16x8;

struct u16x16pair { u16x8 lo, hi; };

// Pack two f32 -> bf16x2: round-half-up + one v_perm_b32 byte-pack (3 VALU).
static __device__ __forceinline__ unsigned pk2(float a, float b) {
  unsigned ua = __float_as_uint(a) + 0x8000u;
  unsigned ub = __float_as_uint(b) + 0x8000u;
#if __has_builtin(__builtin_amdgcn_perm)
  // bytes 0-3 <- second operand, 4-7 <- first; pick high halves of each.
  return __builtin_amdgcn_perm(ub, ua, 0x07060302u);
#else
  return (ua >> 16) | (ub & 0xffff0000u);
#endif
}
static __device__ __forceinline__ u16 f2bf(float f) {
  return (u16)((__float_as_uint(f) + 0x8000u) >> 16);
}

// Build a v16bf WMMA fragment from two 8-element (16B) LDS runs.
static __device__ __forceinline__ v16bf frag16(const u16* p0, const u16* p1) {
  u16x16pair t;
  t.lo = *(const u16x8*)p0;
  t.hi = *(const u16x8*)p1;
  return __builtin_bit_cast(v16bf, t);
}

static __device__ __forceinline__ v8f wmma_bf16(v16bf a, v16bf b, v8f c) {
  // D = A(16x32) * B(32x16) + C(16x16 f32)
  return __builtin_amdgcn_wmma_f32_16x16x32_bf16(false, a, false, b, (short)0, c,
                                                 false, false);
}

// Async global -> LDS copy (16B per lane), tracked by ASYNCcnt.
static __device__ __forceinline__ void async_ld_b128(unsigned lds_off,
                                                     const void* gaddr) {
  asm volatile("global_load_async_to_lds_b128 %0, %1, off"
               :: "v"(lds_off), "v"(gaddr) : "memory");
}
static __device__ __forceinline__ void wait_async0() {
  asm volatile("s_wait_asynccnt 0" ::: "memory");
}
static __device__ __forceinline__ unsigned lds_off32(const void* p) {
  return (unsigned)(unsigned long long)p;   // low 32 bits = LDS byte offset
}

// ---------------------------------------------------------------------------
// Kernel 1: Q = X @ R, K = X @ Ep  (X viewed as (S*B, E)), results stored bf16.
// Block = 256 threads (8 waves). Block tile: 64 rows x 64 cols, for BOTH Q & K
// (waves 0-3 -> Q, waves 4-7 -> K; each wave owns a 16x64 strip = 4 C tiles).
// ---------------------------------------------------------------------------
__global__ __launch_bounds__(256)
void qkproj_kernel(const float* __restrict__ X, const float* __restrict__ R,
                   const float* __restrict__ Ep,
                   u16* __restrict__ qws, u16* __restrict__ kws) {
  __shared__ u16 As [64][32];   // X chunk (rows m, cols k), bf16
  __shared__ u16 BqT[64][32];   // R chunk transposed: [n][k]
  __shared__ u16 BkT[64][32];   // Ep chunk transposed: [n][k]

  const int tid  = threadIdx.x;
  const int lane = tid & 31;
  const int w    = tid >> 5;
  const int lo   = lane & 15;
  const int hi   = lane >> 4;          // lane half
  const int klo  = hi ? 8 : 0;         // A-frag K base (runs klo..+7, klo+16..+7)
  const int kb   = hi ? 16 : 0;        // unused; B-frag uses the same klo runs as A

  const long mbase = (long)blockIdx.x * 64;
  const int  nbase = blockIdx.y * 64;
  const int  mrow  = w & 3;
  const bool isK   = (w >= 4);

  v8f acc[4];
  for (int i = 0; i < 4; ++i) acc[i] = {};

  for (int kc = 0; kc < EQ; kc += 32) {
    __syncthreads();
    // Stage X chunk: 64x32 f32 -> bf16, packed 8B stores (512 float4, 2/thread)
    for (int i = 0; i < 2; ++i) {
      int idx = tid + i * 256;
      int r = idx >> 3, c4 = idx & 7;
      float4 v = *(const float4*)(X + (mbase + r) * EQ + kc + c4 * 4);
      uint2 p; p.x = pk2(v.x, v.y); p.y = pk2(v.z, v.w);
      *(uint2*)&As[r][c4 * 4] = p;
    }
    // Stage R/Ep chunks (32x64) transposed: one k-pair x 4 n per thread,
    // (k, k+1) packed into one b32 LDS store per n.
    {
      int kp = tid >> 4, n4 = tid & 15;          // kp: 0..15, n4: 0..15
      const float* r0 = R  + (size_t)(kc + 2 * kp) * EQ + nbase + n4 * 4;
      const float* e0 = Ep + (size_t)(kc + 2 * kp) * EQ + nbase + n4 * 4;
      float4 ra = *(const float4*)r0;
      float4 rb = *(const float4*)(r0 + EQ);
      float4 ea = *(const float4*)e0;
      float4 eb = *(const float4*)(e0 + EQ);
      *(unsigned*)&BqT[n4*4+0][kp*2] = pk2(ra.x, rb.x);
      *(unsigned*)&BqT[n4*4+1][kp*2] = pk2(ra.y, rb.y);
      *(unsigned*)&BqT[n4*4+2][kp*2] = pk2(ra.z, rb.z);
      *(unsigned*)&BqT[n4*4+3][kp*2] = pk2(ra.w, rb.w);
      *(unsigned*)&BkT[n4*4+0][kp*2] = pk2(ea.x, eb.x);
      *(unsigned*)&BkT[n4*4+1][kp*2] = pk2(ea.y, eb.y);
      *(unsigned*)&BkT[n4*4+2][kp*2] = pk2(ea.z, eb.z);
      *(unsigned*)&BkT[n4*4+3][kp*2] = pk2(ea.w, eb.w);
    }
    __syncthreads();

    const u16* arow = &As[mrow*16 + lo][0];
    v16bf a = frag16(arow + klo, arow + klo + 16);
    auto* Bt = isK ? BkT : BqT;
    v16bf bfr[4];
    for (int nt = 0; nt < 4; ++nt) {
      const u16* brow = &Bt[nt*16 + lo][0];
      bfr[nt] = frag16(brow + klo, brow + klo + 16);
    }
    for (int nt = 0; nt < 4; ++nt)
      acc[nt] = wmma_bf16(a, bfr[nt], acc[nt]);
  }

  u16* dst = isK ? kws : qws;
  for (int nt = 0; nt < 4; ++nt)
    for (int j = 0; j < 8; ++j) {
      long m = mbase + mrow*16 + j + hi*8;     // C layout: VGPR j -> row j (+8 hi)
      int  n = nbase + nt*16 + lo;
      dst[m * EQ + n] = f2bf(acc[nt][j]);
    }
}

// ---------------------------------------------------------------------------
// Kernel 2: per-s attention, flash-style.
// Block = (qt, es, s): 64 query rows (batch axis), 128-wide E slice of output.
// Whole Q tile (64x768 bf16, 96KB) async-staged ONCE into big gfx1250 LDS;
// K chunks double-buffered with async global->LDS overlap (1 barrier/chunk).
// ---------------------------------------------------------------------------
__global__ __launch_bounds__(256)
void attn_kernel(const float* __restrict__ X, const u16* __restrict__ qws,
                 const u16* __restrict__ kws, float* __restrict__ Out) {
  __shared__ u16   Qlds[64][EQ];      // full Q tile, 96KB, resident all block
  __shared__ u16   BkT [2][64][32];   // Kj chunk double buffer (async dest)
  __shared__ float Sblk[64][64];      // raw scores
  __shared__ u16   Pblk[64][64];      // exp(scores - m) bf16 (A of 2nd GEMM)
  __shared__ u16   BvT [128][64];     // Vj transposed: [ecol][key], bf16
  __shared__ float mrun[64], lrun[64], scalef[64];

  const int tid = threadIdx.x, lane = tid & 31, w = tid >> 5;
  const int lo = lane & 15, hi = lane >> 4;
  const int klo = hi ? 8 : 0, kb = hi ? 16 : 0;

  const int qbase = blockIdx.x * 64;    // query tile (batch axis)
  const int ebase = blockIdx.y * 128;   // output E slice
  const int s     = blockIdx.z;

  const u16*   Qs = qws + (size_t)s * BQ * EQ;
  const u16*   Ks = kws + (size_t)s * BQ * EQ;
  const float* Vs = X   + (size_t)s * BQ * EQ;
  float*       Os = Out + (size_t)s * BQ * EQ;

  const int mrow  = w & 3;              // 16-row strip within 64
  const int nhalf = w >> 2;             // 64-col half of the 128-wide slice
  const float sm_scale = 0.03608439182435161f;   // 1/sqrt(768)

  const int r_cp = tid >> 2;            // staging row (0..63)
  const int seg  = tid & 3;             // 16B segment id

  v8f oacc[4];
  for (int i = 0; i < 4; ++i) oacc[i] = {};

  if (tid < 64) { mrun[tid] = -3.0e38f; lrun[tid] = 0.0f; }

  // ---- prologue: async-stage the full Q tile (24 x 16B per thread) ----
  for (int i = 0; i < 24; ++i) {
    int col = (seg + 4 * i) * 8;        // 0..760
    async_ld_b128(lds_off32(&Qlds[r_cp][col]),
                  Qs + (size_t)(qbase + r_cp) * EQ + col);
  }
  // issue first K chunk (global chunk 0 -> buffer 0)
  async_ld_b128(lds_off32(&BkT[0][r_cp][seg*8]),
                Ks + (size_t)r_cp * EQ + seg*8);

  const int NCH = (BQ / 64) * (EQ / 32);   // 192 global K chunks
  int c = 0;                               // global chunk counter

  for (int j = 0; j < BQ; j += 64) {
    // ---- S_blk = Q(64xE) @ Kj^T, K double-buffered with async overlap ----
    v8f sacc[2]; sacc[0] = {}; sacc[1] = {};
    for (int kci = 0; kci < EQ / 32; ++kci, ++c) {
      wait_async0();            // chunk c landed (this wave's async ops done)
      __syncthreads();          // ... and everyone else's
      if (c + 1 < NCH) {        // prefetch next chunk into the other buffer
        int cn = c + 1;
        int jn = (cn / (EQ / 32)) * 64;
        int kn = (cn % (EQ / 32)) * 32;
        async_ld_b128(lds_off32(&BkT[cn & 1][r_cp][seg*8]),
                      Ks + (size_t)(jn + r_cp) * EQ + kn + seg*8);
      }
      const u16* arow = &Qlds[mrow*16 + lo][kci*32];
      v16bf a = frag16(arow + klo, arow + klo + 16);
      auto* Bb = BkT[c & 1];
      v16bf bfr[2];
      for (int t2 = 0; t2 < 2; ++t2) {
        const u16* brow = &Bb[nhalf*32 + t2*16 + lo][0];
        bfr[t2] = frag16(brow + klo, brow + klo + 16);
      }
      for (int t2 = 0; t2 < 2; ++t2)
        sacc[t2] = wmma_bf16(a, bfr[t2], sacc[t2]);
    }
    // Spill scaled scores to LDS
    for (int t2 = 0; t2 < 2; ++t2)
      for (int jj = 0; jj < 8; ++jj)
        Sblk[mrow*16 + jj + hi*8][nhalf*32 + t2*16 + lo] = sacc[t2][jj] * sm_scale;
    __syncthreads();

    // ---- online softmax, one row per thread (vectorized LDS traffic) ----
    if (tid < 64) {
      float mold = mrun[tid], mx = mold;
      for (int c4 = 0; c4 < 16; ++c4) {
        float4 v = *(const float4*)&Sblk[tid][c4*4];
        mx = fmaxf(mx, fmaxf(fmaxf(v.x, v.y), fmaxf(v.z, v.w)));
      }
      float corr = __expf(mold - mx);
      float l = lrun[tid] * corr;
      for (int c4 = 0; c4 < 16; ++c4) {
        float4 v = *(const float4*)&Sblk[tid][c4*4];
        float p0 = __expf(v.x - mx), p1 = __expf(v.y - mx);
        float p2 = __expf(v.z - mx), p3 = __expf(v.w - mx);
        l += (p0 + p1) + (p2 + p3);
        uint2 pq; pq.x = pk2(p0, p1); pq.y = pk2(p2, p3);
        *(uint2*)&Pblk[tid][c4*4] = pq;
      }
      mrun[tid] = mx; lrun[tid] = l; scalef[tid] = corr;
    }
    __syncthreads();

    // ---- stage Vj (64 keys x 128 ecols) transposed, packed k-pair b32 ----
    for (int i = 0; i < 4; ++i) {
      int idx = tid + i * 256;             // 0..1023
      int kp = idx >> 5, n4 = idx & 31;    // kp: key-pair, n4: ecol/4
      const float* v0 = Vs + (size_t)(j + 2*kp) * EQ + ebase + n4 * 4;
      float4 va = *(const float4*)v0;
      float4 vb = *(const float4*)(v0 + EQ);
      *(unsigned*)&BvT[n4*4+0][kp*2] = pk2(va.x, vb.x);
      *(unsigned*)&BvT[n4*4+1][kp*2] = pk2(va.y, vb.y);
      *(unsigned*)&BvT[n4*4+2][kp*2] = pk2(va.z, vb.z);
      *(unsigned*)&BvT[n4*4+3][kp*2] = pk2(va.w, vb.w);
    }
    __syncthreads();

    // ---- rescale accumulators, then O += P(64x64) @ Vj(64x128) ----
    float rs[8];
    for (int jj = 0; jj < 8; ++jj) rs[jj] = scalef[mrow*16 + jj + hi*8];
    for (int nt = 0; nt < 4; ++nt)
      for (int jj = 0; jj < 8; ++jj) oacc[nt][jj] *= rs[jj];
    for (int kc2 = 0; kc2 < 64; kc2 += 32) {
      const u16* arow = &Pblk[mrow*16 + lo][kc2];
      v16bf a = frag16(arow + klo, arow + klo + 16);
      v16bf bfr[4];
      for (int nt = 0; nt < 4; ++nt) {
        const u16* brow = &BvT[nhalf*64 + nt*16 + lo][kc2];
        bfr[nt] = frag16(brow + klo, brow + klo + 16);
      }
      for (int nt = 0; nt < 4; ++nt)
        oacc[nt] = wmma_bf16(a, bfr[nt], oacc[nt]);
    }
  }
  __syncthreads();

  float rl[8];
  for (int jj = 0; jj < 8; ++jj) rl[jj] = 1.0f / lrun[mrow*16 + jj + hi*8];
  for (int nt = 0; nt < 4; ++nt)
    for (int jj = 0; jj < 8; ++jj) {
      int m = qbase + mrow*16 + jj + hi*8;
      int n = ebase + nhalf*64 + nt*16 + lo;
      Os[(size_t)m * EQ + n] = oacc[nt][jj] * rl[jj];
    }
}

extern "C" void kernel_launch(void* const* d_in, const int* in_sizes, int n_in,
                              void* d_out, int out_size, void* d_ws, size_t ws_size,
                              hipStream_t stream) {
  (void)in_sizes; (void)n_in; (void)out_size; (void)ws_size;
  const float* X  = (const float*)d_in[0];   // (S,B,E) f32
  const float* R  = (const float*)d_in[1];   // (E,E) f32
  const float* Ep = (const float*)d_in[2];   // (E,E) f32
  float* Out = (float*)d_out;                // (S,B,E) f32

  // Workspace: Q and K in bf16, S*B*E u16 each (~402 MB each).
  u16* qws = (u16*)d_ws;
  u16* kws = qws + (size_t)SQ * BQ * EQ;

  // Kernel 1: fused Q/K projection. Grid: (S*B/64) x (E/64).
  qkproj_kernel<<<dim3((SQ * BQ) / 64, EQ / 64), 256, 0, stream>>>(X, R, Ep, qws, kws);

  // Kernel 2: fused attention. Grid: (B/64 query tiles) x (E/128 slices) x S.
  attn_kernel<<<dim3(BQ / 64, EQ / 128, SQ), 256, 0, stream>>>(X, qws, kws, Out);
}
